// E80FullRankGateCell_31138512896465
// MI455X (gfx1250) — matrix operand order repaired and checked by reading, not compiled, not run on hardware
//
#include <hip/hip_runtime.h>

// Problem dims (fixed by the reference): T=512, B=64, D=1024, N=64, 4N=256
#define T_DIM 512
#define B_DIM 64
#define D_DIM 1024
#define N_DIM 64
#define FN_DIM 256

typedef __attribute__((ext_vector_type(16))) __bf16 v16bf;
typedef __attribute__((ext_vector_type(8)))  float  v8f;
typedef __attribute__((ext_vector_type(4)))  float  v4f;

// Split a run of 8 consecutive f32 into bf16 hi + lo residual, placing them at
// [base, base+8) of the A/B fragment vectors.
__device__ __forceinline__ void pack8_hilo(const float* __restrict__ p,
                                           v16bf& hi, v16bf& lo, int base) {
  v4f u = *(const v4f*)p;
  v4f w = *(const v4f*)(p + 4);
#pragma unroll
  for (int i = 0; i < 4; ++i) {
    float f = u[i]; __bf16 h = (__bf16)f;
    hi[base + i] = h; lo[base + i] = (__bf16)(f - (float)h);
  }
#pragma unroll
  for (int i = 0; i < 4; ++i) {
    float f = w[i]; __bf16 h = (__bf16)f;
    hi[base + 4 + i] = h; lo[base + 4 + i] = (__bf16)(f - (float)h);
  }
}

// -----------------------------------------------------------------------------
// Phase 0: one-shot pre-pack of W into bf16 hi/lo WMMA B-fragments.
// Fragment f = (k/32)*16 + ct; each fragment = 32 lanes x (16 hi + 16 lo) bf16
// = 1024 bf16 = 2 KB. 512 fragments total = 1 MB hi+lo (vs 1 MB f32 source).
// B-operand lane layout (contiguous k, same order as A): lane = output column
// n = ct*16 + (lane&15); lane-half selects K 0..15 vs 16..31.
// -----------------------------------------------------------------------------
__global__ __launch_bounds__(256) void e80_pack_w(
    const float* __restrict__ W,      // [256, 1024]
    __bf16* __restrict__ Wpk) {       // [512 fragments][1024 bf16]
  const int tid  = blockIdx.x * 256 + threadIdx.x;   // 0..16383
  const int lane = tid & 31;
  const int f    = tid >> 5;                         // fragment 0..511
  const int ct   = f & 15;
  const int ks   = f >> 4;                           // k-step 0..31
  const int n    = ct * 16 + (lane & 15);
  const int kb   = ks * 32 + (lane >> 4) * 16;

  const float* src = W + (long)n * D_DIM + kb;
  v16bf hi, lo;
  pack8_hilo(src,     hi, lo, 0);
  pack8_hilo(src + 8, hi, lo, 8);

  __bf16* dst = Wpk + (long)f * 1024;
  *(v16bf*)(dst + lane * 16)       = hi;
  *(v16bf*)(dst + 512 + lane * 16) = lo;
}

// -----------------------------------------------------------------------------
// Phase 1: proj[T*B, 256] = x[T*B, 1024] @ W[256, 1024]^T   (split-bf16 WMMA)
// Block = 256 threads = 8 waves. Wave w handles rows blockIdx.x*128 + w*16,
// and sweeps all 16 column tiles with 16 f32 accumulators. B fragments come
// pre-packed from L2 (W is resident); only A is converted in-register.
// -----------------------------------------------------------------------------
__global__ __launch_bounds__(256) void e80_proj_gemm(
    const float* __restrict__ x,       // [32768, 1024]
    const __bf16* __restrict__ Wpk,    // packed fragments, see e80_pack_w
    float* __restrict__ proj) {        // [32768, 256]
  const int lane     = threadIdx.x & 31;
  const int wave     = threadIdx.x >> 5;
  const int laneMod  = lane & 15;
  const int laneHalf = lane >> 4;

  const long rowBase = (long)blockIdx.x * 128 + (long)wave * 16;
  const float* xrow  = x + (rowBase + laneMod) * (long)D_DIM;

  v8f acc[16] = {};   // 16x16 f32 tiles covering all 256 output columns

  for (int k = 0; k < D_DIM; k += 32) {
    // ---- A fragment: 16x32 bf16, lane L<16 = row L, K = {0..15};
    //      lane L+16 = row L, K = {16..31}
    const int ka = k + laneHalf * 16;
    v16bf ahi, alo;
    pack8_hilo(xrow + ka,      ahi, alo, 0);
    pack8_hilo(xrow + ka + 8,  ahi, alo, 8);
    if (k + 32 < D_DIM)
      __builtin_prefetch(xrow + ka + 32, 0, 0);   // global_prefetch_b8

    const __bf16* fragRow = Wpk + ((long)(k >> 5) * 16) * 1024 + lane * 16;
#pragma unroll
    for (int ct = 0; ct < 16; ++ct) {
      const __bf16* fp = fragRow + (long)ct * 1024;
      const v16bf bhi = *(const v16bf*)fp;
      const v16bf blo = *(const v16bf*)(fp + 512);

      // acc += Ahi*Bhi + Ahi*Blo + Alo*Bhi  (~fp32 accuracy, f32 accumulate)
      acc[ct] = __builtin_amdgcn_wmma_f32_16x16x32_bf16(
          false, ahi, false, bhi, (short)0, acc[ct], false, false);
      acc[ct] = __builtin_amdgcn_wmma_f32_16x16x32_bf16(
          false, ahi, false, blo, (short)0, acc[ct], false, false);
      acc[ct] = __builtin_amdgcn_wmma_f32_16x16x32_bf16(
          false, alo, false, bhi, (short)0, acc[ct], false, false);
    }
  }

  // D layout: VGPR j -> row laneHalf*8 + j, col = laneMod
#pragma unroll
  for (int ct = 0; ct < 16; ++ct) {
    const long row0 = rowBase + laneHalf * 8;
#pragma unroll
    for (int j = 0; j < 8; ++j)
      proj[(row0 + j) * FN_DIM + ct * 16 + laneMod] = acc[ct][j];
  }
}

// -----------------------------------------------------------------------------
// Phase 2: sequential gated scan. One block per batch b, 64 threads (2 waves);
// thread i owns row i of S and M in VGPRs (128 f32). k/v/q/m broadcast via LDS;
// B_S/B_M staged in LDS with pitch 65 to avoid 64-way bank conflicts.
// -----------------------------------------------------------------------------
__device__ __forceinline__ float sigmoidf_fast(float v) {
  return 1.0f / (1.0f + __expf(-v));
}

__global__ __launch_bounds__(64) void e80_scan(
    const float* __restrict__ proj,   // [T, B, 256]
    const float* __restrict__ S0,     // [B, 64, 64]
    const float* __restrict__ M0,     // [B, 64, 64]
    const float* __restrict__ B_S,    // [64, 64]
    const float* __restrict__ B_M,    // [64, 64]
    float* __restrict__ out,          // [T, B, 64]
    float* __restrict__ S_out,        // [B, 64, 64]
    float* __restrict__ M_out) {      // [B, 64, 64]
  const int b = blockIdx.x;
  const int i = threadIdx.x;          // state row owned by this thread

  __shared__ float sK[N_DIM], sV[N_DIM], sQ[N_DIM], sMv[N_DIM];
  __shared__ float sBS[N_DIM * 65];
  __shared__ float sBM[N_DIM * 65];

  float S[N_DIM], M[N_DIM];
  const size_t stateBase = (size_t)b * N_DIM * N_DIM + (size_t)i * N_DIM;
#pragma unroll
  for (int j = 0; j < N_DIM; ++j) {
    S[j] = S0[stateBase + j];
    M[j] = M0[stateBase + j];
  }
#pragma unroll 8
  for (int j = 0; j < N_DIM; ++j) {
    sBS[i * 65 + j] = B_S[i * N_DIM + j];
    sBM[i * 65 + j] = B_M[i * N_DIM + j];
  }
  __syncthreads();

  for (int t = 0; t < T_DIM; ++t) {
    const float* pr = proj + ((size_t)t * B_DIM + b) * FN_DIM;
    sK[i]  = pr[i];
    sV[i]  = pr[64 + i];
    sQ[i]  = pr[128 + i];
    sMv[i] = pr[192 + i];
    __syncthreads();

    // norms of k and m (redundant per-thread; LDS broadcast reads are cheap)
    float nk = 0.f, nm = 0.f;
#pragma unroll
    for (int j = 0; j < N_DIM; ++j) {
      nk = fmaf(sK[j], sK[j], nk);
      nm = fmaf(sMv[j], sMv[j], nm);
    }
    const float rk = 1.0f / (sqrtf(nk) + 1e-6f);
    const float rm = 1.0f / (sqrtf(nm) + 1e-6f);

    // matvecs against OLD M, OLD S
    float Mk = 0.f, Sk = 0.f, Mm = 0.f;
#pragma unroll
    for (int j = 0; j < N_DIM; ++j) {
      const float knj = sK[j] * rk;
      const float mnj = sMv[j] * rm;
      Mk = fmaf(M[j], knj, Mk);
      Sk = fmaf(S[j], knj, Sk);
      Mm = fmaf(M[j], mnj, Mm);
    }
    const float sd = sV[i] - Sk;           // s_delta[i]

    // S update (gated by old M), fused with S_m and S@q on the NEW S
    float Sm = 0.f, Sq = 0.f;
#pragma unroll
    for (int j = 0; j < N_DIM; ++j) {
      const float knj = sK[j] * rk;
      const float gs  = sigmoidf_fast(M[j] + Mk * knj + sBS[i * 65 + j]);
      const float sn  = fmaf(gs, S[j], sd * knj);
      S[j] = sn;
      const float mnj = sMv[j] * rm;
      Sm = fmaf(sn, mnj, Sm);
      Sq = fmaf(sn, sQ[j], Sq);
    }

    // M update (gated by NEW S); m_delta uses OLD M (Mm captured above)
    const float md = sd - Mm;
#pragma unroll
    for (int j = 0; j < N_DIM; ++j) {
      const float mnj = sMv[j] * rm;
      const float gm  = sigmoidf_fast(S[j] + Sm * mnj + sBM[i * 65 + j]);
      M[j] = fmaf(gm, M[j], md * mnj);
    }

    // out = Sq * silu(Sq) = Sq^2 * sigmoid(Sq)
    out[((size_t)t * B_DIM + b) * N_DIM + i] = Sq * Sq * sigmoidf_fast(Sq);
    __syncthreads();   // protect sK/sV/sQ/sMv before next step overwrites
  }

#pragma unroll
  for (int j = 0; j < N_DIM; ++j) {
    S_out[stateBase + j] = S[j];
    M_out[stateBase + j] = M[j];
  }
}

// -----------------------------------------------------------------------------
extern "C" void kernel_launch(void* const* d_in, const int* in_sizes, int n_in,
                              void* d_out, int out_size, void* d_ws, size_t ws_size,
                              hipStream_t stream) {
  const float* x    = (const float*)d_in[0];   // [T,B,D]
  const float* S0   = (const float*)d_in[1];   // [B,N,N]
  const float* M0   = (const float*)d_in[2];   // [B,N,N]
  const float* W    = (const float*)d_in[3];   // [4N,D]
  const float* B_S  = (const float*)d_in[4];   // [N,N]
  const float* B_M  = (const float*)d_in[5];   // [N,N]

  float* out   = (float*)d_out;                            // [T,B,N]
  float* S_out = out + (size_t)T_DIM * B_DIM * N_DIM;      // [B,N,N]
  float* M_out = S_out + (size_t)B_DIM * N_DIM * N_DIM;    // [B,N,N]

  // Workspace map: proj [T,B,256] f32 (33.55 MB), then packed W (1 MB)
  float*  proj = (float*)d_ws;
  __bf16* Wpk  = (__bf16*)((char*)d_ws +
                           (size_t)T_DIM * B_DIM * FN_DIM * sizeof(float));

  // Phase 0: pack W into WMMA-layout bf16 hi/lo fragments (one-shot, 16K thr)
  e80_pack_w<<<dim3(64), dim3(256), 0, stream>>>(W, Wpk);
  // Phase 1: GEMM, 32768 rows / (8 waves * 16 rows) = 256 workgroups
  e80_proj_gemm<<<dim3(256), dim3(256), 0, stream>>>(x, Wpk, proj);
  // Phase 2: scan, one block per batch chain, state in VGPRs
  e80_scan<<<dim3(B_DIM), dim3(N_DIM), 0, stream>>>(proj, S0, M0, B_S, B_M,
                                                    out, S_out, M_out);
}
